// MLPBlock_17729624998177
// MI455X (gfx1250) — matrix operand order repaired and checked by reading, not compiled, not run on hardware
//
#include <hip/hip_runtime.h>
#include <hip/hip_bf16.h>
#include <math.h>

#define HIDDEN 2880
#define INTER 2880
#define I2 5760
#define NE 32
#define TOPK 4
#define TOKENS 256
#define NTILE 180           // HIDDEN/16 and INTER/16
#define MTOK 32             // tokens per expert tile (2 WMMA N-halves)
#define MAXTILE 8           // ceil(TOKENS/MTOK)
#define MAXSLOTS 64         // sum_e ceil(cnt_e/32) <= 1024/32 + 32 = 64

typedef __attribute__((ext_vector_type(16))) __bf16 v16bf;
typedef __attribute__((ext_vector_type(8)))  float  v8f;

union Frag16 {
    uint4 q[2];
    v16bf v;
};

__device__ __forceinline__ float bfu2f(unsigned short u) {
    unsigned int i = ((unsigned int)u) << 16;
    float f;
    __builtin_memcpy(&f, &i, 4);
    return f;
}
__device__ __forceinline__ unsigned short f2bfu(float f) {
    __hip_bfloat16 h = __float2bfloat16(f);
    unsigned short u;
    __builtin_memcpy(&u, &h, 2);
    return u;
}

// ---------------------------------------------------------------------------
// Kernel 1: fused RMSNorm + gate logits + top-4 + softmax.  One block / token.
// ---------------------------------------------------------------------------
__global__ __launch_bounds__(256)
void moe_rmsnorm_gate(const __hip_bfloat16* __restrict__ x,
                      const float* __restrict__ scale,
                      const __hip_bfloat16* __restrict__ gate_w,
                      const __hip_bfloat16* __restrict__ gate_b,
                      __hip_bfloat16* __restrict__ t,
                      int* __restrict__ topk_idx,
                      float* __restrict__ topk_w) {
    const int tok = blockIdx.x;
    const int tid = threadIdx.x;
    __shared__ float red[256];
    __shared__ unsigned short trow[HIDDEN];
    __shared__ float glog[NE];

    float ss = 0.f;
    for (int i = tid; i < HIDDEN; i += 256) {
        float v = __bfloat162float(x[(size_t)tok * HIDDEN + i]);
        ss += v * v;
    }
    red[tid] = ss;
    __syncthreads();
    for (int s = 128; s > 0; s >>= 1) {
        if (tid < s) red[tid] += red[tid + s];
        __syncthreads();
    }
    const float r = rsqrtf(red[0] / (float)HIDDEN + 1e-5f);

    for (int i = tid; i < HIDDEN; i += 256) {
        float v = __bfloat162float(x[(size_t)tok * HIDDEN + i]) * r * scale[i];
        unsigned short b = f2bfu(v);
        trow[i] = b;
        __hip_bfloat16 hb;
        __builtin_memcpy(&hb, &b, 2);
        t[(size_t)tok * HIDDEN + i] = hb;
    }
    __syncthreads();

    if (tid < NE) {
        const __hip_bfloat16* wr = gate_w + (size_t)tid * HIDDEN;
        float acc = 0.f;
        for (int k = 0; k < HIDDEN; ++k)
            acc += bfu2f(trow[k]) * __bfloat162float(wr[k]);
        acc += __bfloat162float(gate_b[tid]);
        glog[tid] = bfu2f(f2bfu(acc));
    }
    __syncthreads();

    if (tid == 0) {
        float vals[TOPK];
        int idx[TOPK];
        unsigned int used = 0;
        for (int s = 0; s < TOPK; ++s) {
            float best = -1e30f;
            int bi = 0;
            for (int i = 0; i < NE; ++i)
                if (!((used >> i) & 1u) && glog[i] > best) { best = glog[i]; bi = i; }
            used |= (1u << bi);
            vals[s] = best;
            idx[s] = bi;
        }
        float mx = vals[0];
        float e[TOPK], sum = 0.f;
        for (int s = 0; s < TOPK; ++s) { e[s] = expf(vals[s] - mx); sum += e[s]; }
        for (int s = 0; s < TOPK; ++s) {
            topk_idx[tok * TOPK + s] = idx[s];
            topk_w[tok * TOPK + s]   = bfu2f(f2bfu(e[s] / sum));
        }
    }
}

// ---------------------------------------------------------------------------
// Kernel 2: deterministic routing lists + compacted tile-slot prefix sums.
// ---------------------------------------------------------------------------
__global__ __launch_bounds__(32)
void moe_route(const int* __restrict__ topk_idx,
               int* __restrict__ counts,
               int* __restrict__ lists,
               int* __restrict__ tile_base) {
    __shared__ int scnt[NE];
    const int e = threadIdx.x;
    int c = 0;
    for (int tk = 0; tk < TOKENS; ++tk)
        for (int s = 0; s < TOPK; ++s)
            if (topk_idx[tk * TOPK + s] == e)
                lists[e * TOKENS + (c++)] = tk * TOPK + s;   // packed (token, slot)
    counts[e] = c;
    scnt[e] = c;
    __syncthreads();
    if (e == 0) {
        int acc = 0;
        for (int i = 0; i < NE; ++i) {
            tile_base[i] = acc;
            acc += (scnt[i] + MTOK - 1) / MTOK;
        }
    }
}

// ---------------------------------------------------------------------------
// Kernel 3a: GEMM1 + swiglu.  grid=(8 tiles, 32 experts), block=256 (8 wave32)
//   M = 32 tokens per tile: each weight fragment (A) is reused for both token
//   halves (two B fragments / two accumulators) -> halves HBM weight traffic.
//   Output a_scr stored bf16 in [tok32][k] fragment-ready layout.
// ---------------------------------------------------------------------------
__global__ __launch_bounds__(256)
void moe_mlp1(const __hip_bfloat16* __restrict__ t,
              const __hip_bfloat16* __restrict__ mlp1_w,
              const __hip_bfloat16* __restrict__ mlp1_b,
              const int* __restrict__ counts,
              const int* __restrict__ lists,
              const int* __restrict__ tile_base,
              unsigned short* __restrict__ a_scr) {
    __shared__ unsigned short tTs[HIDDEN * MTOK];   // [tok][k]  184320 B
    __shared__ int s_tok[MTOK];

    const int e    = blockIdx.y;
    const int tile = blockIdx.x;
    const int cnt  = counts[e];
    if (tile * MTOK >= cnt) return;
    const int n_tok = min(MTOK, cnt - tile * MTOK);
    const int slot  = tile_base[e] + tile;
    const int tid  = threadIdx.x;
    const int lane = tid & 31;
    const int wave = tid >> 5;

    if (tid < MTOK) {
        int packed = (tid < n_tok) ? lists[e * TOKENS + tile * MTOK + tid]
                                   : lists[e * TOKENS + tile * MTOK];  // pad valid
        s_tok[tid] = packed;
    }
    __syncthreads();

    // stage gathered activation rows into LDS: tTs[tok][k]
    for (int c = tid; c < HIDDEN * MTOK / 4; c += 256) {
        const int k4  = c % (HIDDEN / 4);
        const int tok = c / (HIDDEN / 4);
        const int tokid = s_tok[tok] >> 2;
        uint2 d = ((const uint2*)(t + (size_t)tokid * HIDDEN))[k4];
        const int kbase = k4 * 4;
        tTs[(size_t)tok * HIDDEN + kbase + 0] = (unsigned short)(d.x & 0xffffu);
        tTs[(size_t)tok * HIDDEN + kbase + 1] = (unsigned short)(d.x >> 16);
        tTs[(size_t)tok * HIDDEN + kbase + 2] = (unsigned short)(d.y & 0xffffu);
        tTs[(size_t)tok * HIDDEN + kbase + 3] = (unsigned short)(d.y >> 16);
    }
    __syncthreads();

    const int m  = lane & 15;
    const int kb = (lane < 16) ? 0 : 8;
    const int mo = (lane >= 16) ? 8 : 0;

    for (int nt = wave; nt < NTILE; nt += 8) {
        v8f accg[2] = {{}, {}};
        v8f accl[2] = {{}, {}};
        const __hip_bfloat16* wg = mlp1_w + ((size_t)e * I2 + nt * 16 + m) * HIDDEN + kb;
        const __hip_bfloat16* wl = wg + (size_t)INTER * HIDDEN;
        for (int k0 = 0; k0 < HIDDEN; k0 += 32) {
            Frag16 fa, fl, fb0, fb1;
            fa.q[0] = *(const uint4*)(wg + k0);
            fa.q[1] = *(const uint4*)(wg + k0 + 16);
            fl.q[0] = *(const uint4*)(wl + k0);
            fl.q[1] = *(const uint4*)(wl + k0 + 16);
            const unsigned short* bp = tTs + (size_t)(lane & 15) * HIDDEN + k0 + kb;
            fb0.q[0] = *(const uint4*)(bp);
            fb0.q[1] = *(const uint4*)(bp + 16);
            fb1.q[0] = *(const uint4*)(bp + 16 * HIDDEN);
            fb1.q[1] = *(const uint4*)(bp + 16 * HIDDEN + 16);
            __builtin_prefetch(wg + k0 + 128, 0, 0);
            __builtin_prefetch(wl + k0 + 128, 0, 0);
            accg[0] = __builtin_amdgcn_wmma_f32_16x16x32_bf16(
                false, fa.v, false, fb0.v, (short)0, accg[0], false, false);
            accg[1] = __builtin_amdgcn_wmma_f32_16x16x32_bf16(
                false, fa.v, false, fb1.v, (short)0, accg[1], false, false);
            accl[0] = __builtin_amdgcn_wmma_f32_16x16x32_bf16(
                false, fl.v, false, fb0.v, (short)0, accl[0], false, false);
            accl[1] = __builtin_amdgcn_wmma_f32_16x16x32_bf16(
                false, fl.v, false, fb1.v, (short)0, accl[1], false, false);
        }
        unsigned short* dst = a_scr + (size_t)slot * INTER * MTOK;
        #pragma unroll
        for (int h = 0; h < 2; ++h) {
            #pragma unroll
            for (int rr = 0; rr < 8; ++rr) {
                const int c  = nt * 16 + rr + mo;
                const int n2 = (lane & 15) + 16 * h;
                float bg = __bfloat162float(mlp1_b[(size_t)e * I2 + c]);
                float bl = __bfloat162float(mlp1_b[(size_t)e * I2 + INTER + c]);
                float hg = bfu2f(f2bfu(accg[h][rr] + bg));
                float hl = bfu2f(f2bfu(accl[h][rr] + bl));
                float av = hg * (1.0f / (1.0f + expf(-1.702f * hg))) * (hl + 1.0f);
                dst[(size_t)n2 * INTER + c] = f2bfu(av);
            }
        }
    }
}

// ---------------------------------------------------------------------------
// Kernel 3b: GEMM2.  Stages a_scr tile into LDS with contiguous b128 copies,
//   streams mlp2_w once, reuses each weight fragment for both token halves,
//   scales by routing weight and scatters into the per-(token,slot) buffer.
// ---------------------------------------------------------------------------
__global__ __launch_bounds__(256)
void moe_mlp2(const unsigned short* __restrict__ a_scr,
              const __hip_bfloat16* __restrict__ mlp2_w,
              const __hip_bfloat16* __restrict__ mlp2_b,
              const int* __restrict__ counts,
              const int* __restrict__ lists,
              const int* __restrict__ tile_base,
              const float* __restrict__ topk_w,
              float* __restrict__ slots) {
    __shared__ unsigned short aTs[INTER * MTOK];    // [tok][k]  184320 B
    __shared__ int   s_tok[MTOK];
    __shared__ float s_w[MTOK];

    const int e    = blockIdx.y;
    const int tile = blockIdx.x;
    const int cnt  = counts[e];
    if (tile * MTOK >= cnt) return;
    const int n_tok = min(MTOK, cnt - tile * MTOK);
    const int slot  = tile_base[e] + tile;
    const int tid  = threadIdx.x;
    const int lane = tid & 31;
    const int wave = tid >> 5;

    if (tid < MTOK) {
        int packed = (tid < n_tok) ? lists[e * TOKENS + tile * MTOK + tid]
                                   : lists[e * TOKENS + tile * MTOK];
        s_tok[tid] = packed;
        s_w[tid]   = topk_w[packed];
    }

    // stage swiglu tile (already [tok32][k]) with contiguous 16B copies
    const uint4* src = (const uint4*)(a_scr + (size_t)slot * INTER * MTOK);
    for (int c = tid; c < INTER * MTOK / 8; c += 256)
        ((uint4*)aTs)[c] = src[c];
    __syncthreads();

    const int m  = lane & 15;
    const int kb = (lane < 16) ? 0 : 8;
    const int mo = (lane >= 16) ? 8 : 0;

    for (int ht = wave; ht < NTILE; ht += 8) {
        v8f acc[2] = {{}, {}};
        const __hip_bfloat16* w2 = mlp2_w + ((size_t)e * HIDDEN + ht * 16 + m) * HIDDEN + kb;
        for (int k0 = 0; k0 < INTER; k0 += 32) {
            Frag16 fa, fb0, fb1;
            fa.q[0] = *(const uint4*)(w2 + k0);
            fa.q[1] = *(const uint4*)(w2 + k0 + 16);
            const unsigned short* bp = aTs + (size_t)(lane & 15) * INTER + k0 + kb;
            fb0.q[0] = *(const uint4*)(bp);
            fb0.q[1] = *(const uint4*)(bp + 16);
            fb1.q[0] = *(const uint4*)(bp + 16 * INTER);
            fb1.q[1] = *(const uint4*)(bp + 16 * INTER + 16);
            __builtin_prefetch(w2 + k0 + 128, 0, 0);
            acc[0] = __builtin_amdgcn_wmma_f32_16x16x32_bf16(
                false, fa.v, false, fb0.v, (short)0, acc[0], false, false);
            acc[1] = __builtin_amdgcn_wmma_f32_16x16x32_bf16(
                false, fa.v, false, fb1.v, (short)0, acc[1], false, false);
        }
        #pragma unroll
        for (int h = 0; h < 2; ++h) {
            const int n2 = (lane & 15) + 16 * h;
            if (n2 < n_tok) {
                const float wgt = s_w[n2];
                float* dst = slots + (size_t)s_tok[n2] * HIDDEN;
                #pragma unroll
                for (int rr = 0; rr < 8; ++rr) {
                    const int c = ht * 16 + rr + mo;
                    float o = bfu2f(f2bfu(acc[h][rr] +
                              __bfloat162float(mlp2_b[(size_t)e * HIDDEN + c])));
                    dst[c] = o * wgt;
                }
            }
        }
    }
}

// ---------------------------------------------------------------------------
// Kernel 4: combine residual + weighted expert outputs.
// ---------------------------------------------------------------------------
__global__ __launch_bounds__(256)
void moe_combine(const __hip_bfloat16* __restrict__ x,
                 const float* __restrict__ slots,
                 __hip_bfloat16* __restrict__ out) {
    const int tok = blockIdx.x;
    for (int c = threadIdx.x; c < HIDDEN; c += 256) {
        float acc = __bfloat162float(x[(size_t)tok * HIDDEN + c]);
        #pragma unroll
        for (int s = 0; s < TOPK; ++s)
            acc += slots[((size_t)tok * TOPK + s) * HIDDEN + c];
        out[(size_t)tok * HIDDEN + c] = __float2bfloat16(acc);
    }
}

// ---------------------------------------------------------------------------
extern "C" void kernel_launch(void* const* d_in, const int* in_sizes, int n_in,
                              void* d_out, int out_size, void* d_ws, size_t ws_size,
                              hipStream_t stream) {
    const __hip_bfloat16* x      = (const __hip_bfloat16*)d_in[0];
    const float*          nscale = (const float*)d_in[1];
    const __hip_bfloat16* gate_w = (const __hip_bfloat16*)d_in[2];
    const __hip_bfloat16* gate_b = (const __hip_bfloat16*)d_in[3];
    const __hip_bfloat16* mlp1_w = (const __hip_bfloat16*)d_in[4];
    const __hip_bfloat16* mlp1_b = (const __hip_bfloat16*)d_in[5];
    const __hip_bfloat16* mlp2_w = (const __hip_bfloat16*)d_in[6];
    const __hip_bfloat16* mlp2_b = (const __hip_bfloat16*)d_in[7];
    __hip_bfloat16* out = (__hip_bfloat16*)d_out;

    char* ws = (char*)d_ws;
    __hip_bfloat16* t     = (__hip_bfloat16*)(ws);            // 1,474,560 B
    int*   topk_idx       = (int*)  (ws + 1474560);           //     4,096 B
    float* topk_w         = (float*)(ws + 1478656);           //     4,096 B
    int*   counts         = (int*)  (ws + 1482752);           //       128 B
    int*   tile_base      = (int*)  (ws + 1482880);           //       128 B
    int*   lists          = (int*)  (ws + 1483008);           //    32,768 B
    unsigned short* a_scr = (unsigned short*)(ws + 1515776);  // 11,796,480 B
    float* slots          = (float*)(ws + 13312256);          // 11,796,480 B

    moe_rmsnorm_gate<<<TOKENS, 256, 0, stream>>>(x, nscale, gate_w, gate_b,
                                                 t, topk_idx, topk_w);
    moe_route<<<1, 32, 0, stream>>>(topk_idx, counts, lists, tile_base);
    moe_mlp1<<<dim3(MAXTILE, NE), 256, 0, stream>>>(t, mlp1_w, mlp1_b,
                                                    counts, lists, tile_base, a_scr);
    moe_mlp2<<<dim3(MAXTILE, NE), 256, 0, stream>>>(a_scr, mlp2_w, mlp2_b,
                                                    counts, lists, tile_base,
                                                    topk_w, slots);
    moe_combine<<<TOKENS, 256, 0, stream>>>(x, slots, out);
}
